// VectorGnn_7911329760072
// MI455X (gfx1250) — matrix operand order repaired and checked by reading, not compiled, not run on hardware
//
#include <hip/hip_runtime.h>

// ---------------------------------------------------------------------------
// MI455X (gfx1250, wave32) bf16-WMMA masked attention.
//   q = x@Wq+bq ; k = x@Wk+bk ; v = x@Wv+bv
//   out = softmax(q k^T / sqrt(D)) @ v * mask   (row-constant bias cancels)
// B=32, P=2048, C=256, D=128.
// Kernels: weight transpose -> fused qkv projection (x staged once in LDS)
//          -> flash attention (async double-buffered K/V tiles in LDS).
// ---------------------------------------------------------------------------

typedef __attribute__((ext_vector_type(16))) __bf16        v16bf;
typedef __attribute__((ext_vector_type(8)))  float         v8f;
typedef __attribute__((ext_vector_type(8)))  unsigned int  v8u;
typedef __attribute__((ext_vector_type(4)))  int           v4i;

#define BATCH 32
#define SEQ   2048
#define CDIM  256
#define DDIM  128

#if defined(__AMDGCN__) && __has_builtin(__builtin_amdgcn_global_load_async_to_lds_b128)
#define HAS_ASYNC 1
#else
#define HAS_ASYNC 0
#endif

static __device__ __forceinline__ v8f wmma_bf16(v8u a, v8u b, v8f c) {
  return __builtin_amdgcn_wmma_f32_16x16x32_bf16(
      false, __builtin_bit_cast(v16bf, a),
      false, __builtin_bit_cast(v16bf, b),
      (short)0, c, false, false);
}

// f32 -> bf16 via native conversion (hardware cvt, not bit-twiddled VALU)
static __device__ __forceinline__ unsigned short f2bf(float f) {
  return __builtin_bit_cast(unsigned short, (__bf16)f);
}
static __device__ __forceinline__ unsigned pack2bf(float lo, float hi) {
  return (unsigned)f2bf(lo) | ((unsigned)f2bf(hi) << 16);
}

// A-operand (16x32 bf16) dword j K-offset for this lane (ISA 7.12.2 layout)
static __device__ __forceinline__ int a_koff(int j, int half) {
  return 2 * j + ((j >= 4) ? 8 : 0) + (half ? 8 : 0);
}
// B-operand (32x16 bf16) dword j K-offset for this lane
static __device__ __forceinline__ int b_koff(int j, int half) {
  return 2 * j + ((j >= 4) ? 8 : 0) + (half ? 8 : 0);
}

// 16B global -> LDS copy: ASYNCcnt-tracked global_load_async_to_lds_b128.
// Builtin params are typed int4 pointers in device/shared address spaces.
static __device__ __forceinline__ void copy_b128(const unsigned short* g,
                                                 unsigned short* l) {
#if HAS_ASYNC
  typedef __attribute__((address_space(1))) v4i as1_v4i;
  typedef __attribute__((address_space(3))) v4i as3_v4i;
  __builtin_amdgcn_global_load_async_to_lds_b128((as1_v4i*)g, (as3_v4i*)l,
                                                 0, 0);
#else
  *(uint4*)l = *(const uint4*)g;
#endif
}
static __device__ __forceinline__ void async_wait_pend6() {
#if HAS_ASYNC
#if __has_builtin(__builtin_amdgcn_s_wait_asynccnt)
  __builtin_amdgcn_s_wait_asynccnt(6);
#else
  asm volatile("s_wait_asynccnt 0x6" ::: "memory");
#endif
#endif
}
static __device__ __forceinline__ void async_wait_all() {
#if HAS_ASYNC
#if __has_builtin(__builtin_amdgcn_s_wait_asynccnt)
  __builtin_amdgcn_s_wait_asynccnt(0);
#else
  asm volatile("s_wait_asynccnt 0x0" ::: "memory");
#endif
#endif
}

// ---------------------------------------------------------------------------
// Weight transpose + f32->bf16:  W[256,outDim] -> Wt[outDim,256] (bf16 bits)
// ---------------------------------------------------------------------------
__global__ void wtrans_kernel(const float* __restrict__ W,
                              unsigned short* __restrict__ Wt, int outDim) {
  int idx = blockIdx.x * 256 + threadIdx.x;
  int o = idx / 256, c = idx % 256;
  if (o < outDim) Wt[idx] = f2bf(W[(size_t)c * outDim + o]);
}

// ---------------------------------------------------------------------------
// Fused qkv projection. One block = one 128-row x slab, staged ONCE in LDS as
// bf16 (64 KB). Each wave keeps its 16x256 A-operand in VGPRs and sweeps all
// 32 output column tiles (q:8, k:8, v:16). v stored transposed [B,C,P] so the
// P*V B-operand pairs are contiguous later. x HBM traffic: exactly 64 MB.
// ---------------------------------------------------------------------------
__global__ __launch_bounds__(256) void proj_fused_kernel(
    const float* __restrict__ x,
    const unsigned short* __restrict__ wqt, const float* __restrict__ bq,
    const unsigned short* __restrict__ wkt, const float* __restrict__ bk,
    const unsigned short* __restrict__ wvt, const float* __restrict__ bv,
    unsigned short* __restrict__ qout, unsigned short* __restrict__ kout,
    unsigned short* __restrict__ vout) {
  __shared__ unsigned short lds_x[128 * CDIM];  // 64 KB bf16 slab

  const int lane = threadIdx.x & 31;
  const int wave = threadIdx.x >> 5;
  const int half = lane >> 4;
  const int lrow = lane & 15;
  const int row0 = blockIdx.x * 128;

  // ---- stage + convert slab: thread -> half row (128 floats) ----
  {
    const int r  = threadIdx.x >> 1;
    const int c0 = (threadIdx.x & 1) * 128;
    const float*    src = x + (size_t)(row0 + r) * CDIM + c0;
    unsigned short* dst = lds_x + r * CDIM + c0;
#pragma unroll
    for (int i = 0; i < 32; ++i) {
      float4 f = ((const float4*)src)[i];
      uint2 u; u.x = pack2bf(f.x, f.y); u.y = pack2bf(f.z, f.w);
      ((uint2*)dst)[i] = u;
    }
  }
  __syncthreads();

  // ---- wave-resident A operand: 16 rows x 256 K, 8 chunks ----
  v8u ax[8];
  {
    const unsigned short* xrow = lds_x + (wave * 16 + lrow) * CDIM;
#pragma unroll
    for (int kk = 0; kk < 8; ++kk)
#pragma unroll
      for (int j = 0; j < 8; ++j)
        ax[kk][j] = *(const unsigned*)(xrow + kk * 32 + a_koff(j, half));
  }

  const int gr0 = row0 + wave * 16;  // global row of this wave's strip
  for (int seg = 0; seg < 32; ++seg) {
    const unsigned short* Wt;
    const float* bias;
    int ct;
    if (seg < 8)       { Wt = wqt; bias = bq; ct = seg; }
    else if (seg < 16) { Wt = wkt; bias = bk; ct = seg - 8; }
    else               { Wt = wvt; bias = bv; ct = seg - 16; }
    const int col0 = ct * 16;
    const unsigned short* wcol = Wt + (size_t)(col0 + lrow) * CDIM;

    v8f acc = {0.f, 0.f, 0.f, 0.f, 0.f, 0.f, 0.f, 0.f};
#pragma unroll
    for (int kk = 0; kk < 8; ++kk) {
      v8u bu;
#pragma unroll
      for (int j = 0; j < 8; ++j)
        bu[j] = *(const unsigned*)(wcol + kk * 32 + b_koff(j, half));
      acc = wmma_bf16(ax[kk], bu, acc);
    }

    const float bval = bias[col0 + lrow];
    if (seg < 16) {
      unsigned short* o = (seg < 8) ? qout : kout;
#pragma unroll
      for (int g = 0; g < 8; ++g)
        o[(size_t)(gr0 + g + 8 * half) * DDIM + col0 + lrow] =
            f2bf(acc[g] + bval);
    } else {
      const int b = gr0 >> 11, p0 = gr0 & (SEQ - 1);
      unsigned short* vb =
          vout + (size_t)b * CDIM * SEQ + (size_t)(col0 + lrow) * SEQ;
#pragma unroll
      for (int g = 0; g < 8; ++g) vb[p0 + g + 8 * half] = f2bf(acc[g] + bval);
    }
  }
}

// ---------------------------------------------------------------------------
// Flash attention. 8 waves/block share one batch; each wave owns a 16-query
// strip (Q resident in VGPRs). 32-key tiles stream through double-buffered
// LDS via async global->LDS copies (6 b128/thread/tile, ASYNCcnt-tracked).
// Online softmax in f32; P*V accumulated over 16 channel tiles (C=256).
// ---------------------------------------------------------------------------
__global__ __launch_bounds__(256) void flash_attn_kernel(
    const unsigned short* __restrict__ qw,   // [B*P, 128] bf16
    const unsigned short* __restrict__ kw,   // [B*P, 128] bf16
    const unsigned short* __restrict__ vt,   // [B, 256, 2048] bf16
    const float* __restrict__ mask,          // [B, P]
    float* __restrict__ out) {               // [B, P, 256]
  __shared__ unsigned short lds_k[2][32 * DDIM];   // 2 x 8 KB
  __shared__ unsigned short lds_v[2][CDIM * 32];   // 2 x 16 KB
  __shared__ unsigned short lds_p[8][16 * 32];     // 8 KB per-wave P staging

  const int lane  = threadIdx.x & 31;
  const int wave  = threadIdx.x >> 5;
  const int half  = lane >> 4;
  const int lrow  = lane & 15;
  const int batch = blockIdx.x >> 4;
  const int qbase = ((blockIdx.x & 15) * 8 + wave) * 16;

  const unsigned short* kbatch = kw + (size_t)batch * SEQ * DDIM;
  const unsigned short* vbatch = vt + (size_t)batch * CDIM * SEQ;

  // Q strip (16 x 128) -> 4 A-operand chunks, resident in VGPRs
  v8u aq[4];
  {
    const unsigned short* qrow =
        qw + ((size_t)batch * SEQ + qbase + lrow) * DDIM;
#pragma unroll
    for (int kk = 0; kk < 4; ++kk)
#pragma unroll
      for (int j = 0; j < 8; ++j)
        aq[kk][j] = *(const unsigned*)(qrow + kk * 32 + a_koff(j, half));
  }

  v8f O[16];
#pragma unroll
  for (int t = 0; t < 16; ++t) O[t] = (v8f){0.f,0.f,0.f,0.f,0.f,0.f,0.f,0.f};
  float rm[8], rs[8];
#pragma unroll
  for (int g = 0; g < 8; ++g) { rm[g] = -3.0e38f; rs[g] = 0.f; }

  const float sl2e = 0.08838834764831845f * 1.4426950408889634f; // scale*log2e
  const int NT = SEQ / 32;

  // issue one 32-key tile's K (2 x b128) + V (4 x b128) copies per thread
  auto issue_tile = [&](int kt, int buf) {
    const unsigned short* ksrc = kbatch + (size_t)kt * 32 * DDIM;
#pragma unroll
    for (int i = 0; i < 2; ++i)
      copy_b128(ksrc + (size_t)(threadIdx.x + i * 256) * 8,
                &lds_k[buf][(threadIdx.x + i * 256) * 8]);
    const unsigned short* vsrc = vbatch + (size_t)threadIdx.x * SEQ + kt * 32;
#pragma unroll
    for (int i = 0; i < 4; ++i)
      copy_b128(vsrc + i * 8, &lds_v[buf][threadIdx.x * 32 + i * 8]);
  };

  issue_tile(0, 0);
  for (int kt = 0; kt < NT; ++kt) {
    const int cur = kt & 1;
    __syncthreads();  // all waves done reading buf[1-cur] (tile kt-1)
    if (kt + 1 < NT) {
      issue_tile(kt + 1, 1 - cur);
      async_wait_pend6();  // tile kt's 6 copies complete; kt+1's in flight
    } else {
      async_wait_all();
    }
    __syncthreads();  // every wave's tile-kt copies have landed

    // ---- S = Q K^T : two 16-key subtiles, 4 WMMAs each over D=128 ----
    v8f S[2];
#pragma unroll
    for (int s = 0; s < 2; ++s) {
      v8f c = {0.f,0.f,0.f,0.f,0.f,0.f,0.f,0.f};
#pragma unroll
      for (int kk = 0; kk < 4; ++kk) {
        v8u bu;
#pragma unroll
        for (int j = 0; j < 8; ++j)
          bu[j] = *(const unsigned*)&lds_k[cur][(s * 16 + lrow) * DDIM +
                                               kk * 32 + b_koff(j, half)];
        c = wmma_bf16(aq[kk], bu, c);
      }
      S[s] = c;
    }

    // ---- online softmax (log2 domain); rows live at g + 8*half ----
    float corr[8];
#pragma unroll
    for (int g = 0; g < 8; ++g) {
      const float z0 = S[0][g] * sl2e;
      const float z1 = S[1][g] * sl2e;
      S[0][g] = z0; S[1][g] = z1;
      float m = fmaxf(z0, z1);
      m = fmaxf(m, __shfl_xor(m, 1, 32));
      m = fmaxf(m, __shfl_xor(m, 2, 32));
      m = fmaxf(m, __shfl_xor(m, 4, 32));
      m = fmaxf(m, __shfl_xor(m, 8, 32));
      const float nm = fmaxf(rm[g], m);
      corr[g] = exp2f(rm[g] - nm);
      rm[g] = nm;
    }
    unsigned short* pst = lds_p[wave];
#pragma unroll
    for (int g = 0; g < 8; ++g) {
      const float p0 = exp2f(S[0][g] - rm[g]);
      const float p1 = exp2f(S[1][g] - rm[g]);
      const int r = g + 8 * half;
      pst[r * 32 + lrow]      = f2bf(p0);  // C/D -> A re-layout via LDS
      pst[r * 32 + 16 + lrow] = f2bf(p1);
      float ps = p0 + p1;
      ps += __shfl_xor(ps, 1, 32);
      ps += __shfl_xor(ps, 2, 32);
      ps += __shfl_xor(ps, 4, 32);
      ps += __shfl_xor(ps, 8, 32);
      rs[g] = rs[g] * corr[g] + ps;
    }
#pragma unroll
    for (int t = 0; t < 16; ++t)
#pragma unroll
      for (int g = 0; g < 8; ++g) O[t][g] *= corr[g];

    asm volatile("s_wait_dscnt 0x0" ::: "memory");  // cross-lane LDS re-read

    v8u ap;
#pragma unroll
    for (int j = 0; j < 8; ++j)
      ap[j] = *(const unsigned*)&pst[lrow * 32 + a_koff(j, half)];

    // ---- O += P (16x32) @ V (32x16) over 16 channel tiles ----
#pragma unroll
    for (int t = 0; t < 16; ++t) {
      v8u bu;
#pragma unroll
      for (int j = 0; j < 8; ++j)
        bu[j] = *(const unsigned*)&lds_v[cur][(t * 16 + lrow) * 32 +
                                             b_koff(j, half)];
      O[t] = wmma_bf16(ap, bu, O[t]);
    }
  }

  // ---- epilogue: normalize, apply query-row mask, store f32 ----
#pragma unroll
  for (int g = 0; g < 8; ++g) {
    const int r = qbase + g + 8 * half;
    const float inv = mask[(size_t)batch * SEQ + r] / rs[g];
    float* orow = out + ((size_t)batch * SEQ + r) * CDIM;
#pragma unroll
    for (int t = 0; t < 16; ++t) orow[t * 16 + lrow] = O[t][g] * inv;
  }
}

// ---------------------------------------------------------------------------
extern "C" void kernel_launch(void* const* d_in, const int* in_sizes, int n_in,
                              void* d_out, int out_size, void* d_ws,
                              size_t ws_size, hipStream_t stream) {
  (void)in_sizes; (void)n_in; (void)out_size; (void)ws_size;
  const float* x    = (const float*)d_in[0];
  const float* mask = (const float*)d_in[1];
  const float* Wq   = (const float*)d_in[2];
  const float* bq   = (const float*)d_in[3];
  const float* Wk   = (const float*)d_in[4];
  const float* bk   = (const float*)d_in[5];
  const float* Wv   = (const float*)d_in[6];
  const float* bv   = (const float*)d_in[7];
  float* out = (float*)d_out;

  unsigned short* ws  = (unsigned short*)d_ws;
  unsigned short* qw  = ws;                          // 16 MB
  unsigned short* kw  = ws + (size_t)8388608;        // 16 MB
  unsigned short* vtw = ws + (size_t)16777216;       // 32 MB ([B,C,P])
  unsigned short* wqt = ws + (size_t)33554432;       // 64 KB
  unsigned short* wkt = wqt + 32768;                 // 64 KB
  unsigned short* wvt = wkt + 32768;                 // 128 KB

  wtrans_kernel<<<128, 256, 0, stream>>>(Wq, wqt, DDIM);
  wtrans_kernel<<<128, 256, 0, stream>>>(Wk, wkt, DDIM);
  wtrans_kernel<<<256, 256, 0, stream>>>(Wv, wvt, CDIM);

  proj_fused_kernel<<<BATCH * SEQ / 128, 256, 0, stream>>>(
      x, wqt, bq, wkt, bk, wvt, bv, qw, kw, vtw);

  flash_attn_kernel<<<BATCH * (SEQ / 16) / 8, 256, 0, stream>>>(qw, kw, vtw,
                                                                mask, out);
}
